// LSTM_encoder_layer_18030272708613
// MI455X (gfx1250) — matrix operand order repaired and checked by reading, not compiled, not run on hardware
//
#include <hip/hip_runtime.h>

// ---------------- types ----------------
typedef __bf16 bf16_t;
typedef bf16_t v16bf __attribute__((ext_vector_type(16)));
typedef float  v8f   __attribute__((ext_vector_type(8)));

struct alignas(16) U4 { unsigned int x, y, z, w; };
struct alignas(16) F4 { float x, y, z, w; };

union Frag {
    U4    q[2];
    v16bf v;
};

// ---------------- helpers ----------------
__device__ __forceinline__ unsigned int pack_bf2(float lo, float hi) {
    unsigned int a = __float_as_uint(lo);
    a = (a + 0x7FFFu + ((a >> 16) & 1u)) >> 16;      // RNE f32 -> bf16
    unsigned int b = __float_as_uint(hi);
    b = (b + 0x7FFFu + ((b >> 16) & 1u)) >> 16;
    return a | (b << 16);
}

__device__ __forceinline__ unsigned short f2bf(float f) {
    unsigned int a = __float_as_uint(f);
    return (unsigned short)((a + 0x7FFFu + ((a >> 16) & 1u)) >> 16);
}

__device__ __forceinline__ float fast_sigmoid(float x) {
    return __builtin_amdgcn_rcpf(1.0f + __expf(-x));
}
__device__ __forceinline__ float fast_tanh(float x) {
    float cx = fminf(fmaxf(x, -12.0f), 12.0f);
    float e  = __expf(2.0f * cx);
    return (e - 1.0f) * __builtin_amdgcn_rcpf(e + 1.0f);
}

// ---------------- weight pre-swizzle: f32 [2D,D] -> bf16 B-fragment order ----------------
// Fragment (g, kt, nt) = 256 dwords. lane = n + 16*khalf; vgpr v holds packed
// bf16 pair (K = khalf*16 + 2v, K+1) for column n.
__global__ void swizzle_w(const float* __restrict__ wu, const float* __restrict__ wf,
                          const float* __restrict__ wo, const float* __restrict__ wc,
                          unsigned int* __restrict__ dst, int Dh)
{
    const int KT = (2 * Dh) / 32;
    const int NT = Dh / 16;
    const int total = 4 * KT * NT * 256;
    for (int o = blockIdx.x * blockDim.x + threadIdx.x; o < total;
         o += gridDim.x * blockDim.x) {
        int v    = o & 7;
        int lane = (o >> 3) & 31;
        int f    = o >> 8;
        int nt = f % NT;
        int kt = (f / NT) % KT;
        int g  = f / (NT * KT);
        int khalf = lane >> 4;
        int n     = lane & 15;
        int k0  = kt * 32 + khalf * 16 + 2 * v;
        int col = nt * 16 + n;
        const float* w = (g == 0) ? wu : (g == 1) ? wf : (g == 2) ? wo : wc;
        float lo = w[(size_t)k0 * Dh + col];
        float hi = w[(size_t)(k0 + 1) * Dh + col];
        dst[o] = pack_bf2(lo, hi);
    }
}

// ---------------- fused persistent LSTM ----------------
// One WG = 16 batch rows, full hidden dim; wave wv owns N-tile [wv*16, wv*16+16).
// LDS row layout (bf16): [ x_buf0 (D) | x_buf1 (D) | h_buf0 (D) | h_buf1 (D) | pad(8) ]
// Double-buffered x and h -> ONE barrier per timestep; x_{t+1} global load is
// issued right after the barrier and consumed after the WMMA block.
template <int D, int T>
__global__ __launch_bounds__((D / 16) * 32)
void lstm_fused(const float* __restrict__ x,            // [B, T, D] f32
                const unsigned int* __restrict__ wsw,   // swizzled bf16 weights
                const float* __restrict__ bu, const float* __restrict__ bfv,
                const float* __restrict__ bo, const float* __restrict__ bc,
                float* __restrict__ out_h, float* __restrict__ out_m)
{
    constexpr int NT = D / 16;            // N tiles == waves per WG
    constexpr int KT = (2 * D) / 32;      // K tiles per step
    constexpr int S  = 4 * D + 8;         // LDS row stride: 16B aligned + 4-bank stagger
    constexpr int GS = KT * NT * 256;     // per-gate dword stride in wsw

    alignas(16) __shared__ unsigned short inp[16 * S];

    const int tid   = threadIdx.x;
    const int lane  = tid & 31;
    const int wv    = tid >> 5;
    const int nloc  = lane & 15;
    const int khalf = lane >> 4;
    const int ncol  = wv * 16 + nloc;     // hidden column owned by this lane
    const int bbase = blockIdx.x * 16;

    // zero both h buffers: cols [2D, 4D)
    for (int i = tid; i < 16 * 2 * D; i += NT * 32) {
        int r = i / (2 * D), c = i % (2 * D);
        inp[r * S + 2 * D + c] = 0;
    }

    const float Bu = bu[ncol], Bf = bfv[ncol], Bo = bo[ncol], Bc = bc[ncol];

    v8f m, hf;
#pragma unroll
    for (int i = 0; i < 8; ++i) { m[i] = 0.0f; hf[i] = 0.0f; }

    // x staging: 8 f32 per thread, one row = D/8 threads
    constexpr int TPR = D / 8;
    const int srow = tid / TPR;
    const int scol = (tid % TPR) * 8;
    const float* xrow = x + ((size_t)(bbase + srow) * T) * D + scol;

    // prologue: stage x_0 into x_buf0
    {
        F4 a0 = *(const F4*)xrow;
        F4 a1 = *(const F4*)(xrow + 4);
        U4 pk;
        pk.x = pack_bf2(a0.x, a0.y);
        pk.y = pack_bf2(a0.z, a0.w);
        pk.z = pack_bf2(a1.x, a1.y);
        pk.w = pack_bf2(a1.z, a1.w);
        *(U4*)&inp[srow * S + scol] = pk;
    }

    const unsigned short* aBase = &inp[nloc * S];   // A-fragment row = lane%16

    for (int t = 0; t < T; ++t) {
        __syncthreads();          // buffers (t&1) fully visible
        const int buf  = t & 1;
        const int nbuf = buf ^ 1;

        // ---- issue x_{t+1} global load now; consumed only after compute ----
        const int tn = (t + 1 < T) ? (t + 1) : t;       // clamped (last iter redundant)
        const float* px = xrow + (size_t)tn * D;
        F4 n0 = *(const F4*)px;
        F4 n1 = *(const F4*)(px + 4);

        // ---- accumulators start at the bias (D = A*B + C) ----
        v8f au, af, ao, ac;
#pragma unroll
        for (int i = 0; i < 8; ++i) { au[i] = Bu; af[i] = Bf; ao[i] = Bo; ac[i] = Bc; }

        const int xoff = buf * D;            // current x buffer
        const int hoff = 2 * D + buf * D;    // current h buffer

        // A-fragment LDS element offset for K-tile kt (kt known at unroll time)
        auto aoffs = [&](int kt) {
            int c = kt * 32;
            return (c < D ? xoff + c : hoff + (c - D)) + khalf * 16;
        };

        // 2-deep pipelined A-fragment loads (2x ds_load_b128 each)
        Frag A;
        {
            const unsigned short* pA = aBase + aoffs(0);
            A.q[0] = *(const U4*)pA;
            A.q[1] = *(const U4*)(pA + 8);
        }
#pragma unroll
        for (int kt = 0; kt < KT; ++kt) {
            Frag An = A;
            if (kt + 1 < KT) {
                const unsigned short* pA = aBase + aoffs(kt + 1);
                An.q[0] = *(const U4*)pA;
                An.q[1] = *(const U4*)(pA + 8);
            }

            const unsigned int* pB =
                wsw + (((size_t)kt * NT + wv) << 8) + ((size_t)lane << 3);
            Frag B0, B1, B2, B3;
            B0.q[0] = *(const U4*)(pB);              B0.q[1] = *(const U4*)(pB + 4);
            B1.q[0] = *(const U4*)(pB + GS);         B1.q[1] = *(const U4*)(pB + GS + 4);
            B2.q[0] = *(const U4*)(pB + 2 * GS);     B2.q[1] = *(const U4*)(pB + 2 * GS + 4);
            B3.q[0] = *(const U4*)(pB + 3 * GS);     B3.q[1] = *(const U4*)(pB + 3 * GS + 4);

            au = __builtin_amdgcn_wmma_f32_16x16x32_bf16(false, A.v, false, B0.v, (short)0, au, false, false);
            af = __builtin_amdgcn_wmma_f32_16x16x32_bf16(false, A.v, false, B1.v, (short)0, af, false, false);
            ao = __builtin_amdgcn_wmma_f32_16x16x32_bf16(false, A.v, false, B2.v, (short)0, ao, false, false);
            ac = __builtin_amdgcn_wmma_f32_16x16x32_bf16(false, A.v, false, B3.v, (short)0, ac, false, false);

            A = An;
        }

        // ---- gating (C layout: vgpr i -> row i + 8*khalf, lane -> column) ----
#pragma unroll
        for (int i = 0; i < 8; ++i) {
            float gu = fast_sigmoid(au[i]);
            float gf = fast_sigmoid(af[i]);
            float go = fast_sigmoid(ao[i]);
            float gc = fast_tanh(ac[i]);
            float mv = gf * m[i] + gu * gc;
            m[i]  = mv;
            hf[i] = fast_tanh(go * mv);   // nonstandard: h = tanh(go * m)
        }

        // ---- h_t -> h_buf[nbuf] (read by step t+1) ----
#pragma unroll
        for (int i = 0; i < 8; ++i) {
            int r = i + khalf * 8;
            inp[r * S + 2 * D + nbuf * D + ncol] = f2bf(hf[i]);
        }

        // ---- x_{t+1} (loaded above, latency hidden) -> x_buf[nbuf] ----
        U4 pk;
        pk.x = pack_bf2(n0.x, n0.y);
        pk.y = pack_bf2(n0.z, n0.w);
        pk.z = pack_bf2(n1.x, n1.y);
        pk.w = pack_bf2(n1.z, n1.w);
        *(U4*)&inp[srow * S + nbuf * D + scol] = pk;
    }

    // ---- final outputs (f32) straight from registers ----
#pragma unroll
    for (int i = 0; i < 8; ++i) {
        int r = i + khalf * 8;
        size_t b = (size_t)(bbase + r);
        out_h[b * D + ncol] = hf[i];
        out_m[b * D + ncol] = m[i];
    }
}

// ---------------- launch ----------------
extern "C" void kernel_launch(void* const* d_in, const int* in_sizes, int n_in,
                              void* d_out, int out_size, void* d_ws, size_t ws_size,
                              hipStream_t stream) {
    (void)in_sizes; (void)n_in; (void)out_size; (void)ws_size;

    const float* x0  = (const float*)d_in[0];
    const float* x1  = (const float*)d_in[1];
    const float* wu0 = (const float*)d_in[2];
    const float* wf0 = (const float*)d_in[3];
    const float* wo0 = (const float*)d_in[4];
    const float* wc0 = (const float*)d_in[5];
    const float* bu0 = (const float*)d_in[6];
    const float* bf0 = (const float*)d_in[7];
    const float* bo0 = (const float*)d_in[8];
    const float* bc0 = (const float*)d_in[9];
    const float* wu1 = (const float*)d_in[10];
    const float* wf1 = (const float*)d_in[11];
    const float* wo1 = (const float*)d_in[12];
    const float* wc1 = (const float*)d_in[13];
    const float* bu1 = (const float*)d_in[14];
    const float* bf1 = (const float*)d_in[15];
    const float* bo1 = (const float*)d_in[16];
    const float* bc1 = (const float*)d_in[17];

    // workspace: swizzled bf16 weights (LSTM0: 64K dwords, LSTM1: 256K dwords)
    unsigned int* w0s = (unsigned int*)d_ws;
    unsigned int* w1s = w0s + (4 * 256 * 128) / 2;

    swizzle_w<<<64, 256, 0, stream>>>(wu0, wf0, wo0, wc0, w0s, 128);
    swizzle_w<<<128, 256, 0, stream>>>(wu1, wf1, wo1, wc1, w1s, 256);

    // output tuple layout: hh0[512*128], hh1[512*256], mm0[512*128], mm1[512*256]
    float* hh0 = (float*)d_out;
    float* hh1 = hh0 + 512 * 128;
    float* mm0 = hh1 + 512 * 256;
    float* mm1 = mm0 + 512 * 128;

    lstm_fused<128, 500><<<32, (128 / 16) * 32, 0, stream>>>(x0, w0s, bu0, bf0, bo0, bc0, hh0, mm0);
    lstm_fused<256, 250><<<32, (256 / 16) * 32, 0, stream>>>(x1, w1s, bu1, bf1, bo1, bc1, hh1, mm1);
}
